// LSTM_18708877541422
// MI455X (gfx1250) — matrix operand order repaired and checked by reading, not compiled, not run on hardware
//
#include <hip/hip_runtime.h>

// -------- problem dimensions --------
#define B_    64
#define S_    512
#define D_    256
#define H_    1024
#define O_    128
#define K_    (H_ + D_)          // 1280: combined GEMM K = [h | x]
#define KT_   (K_ / 32)          // 40 wmma K-steps of 32
#define KTH_  (H_ / 32)          // first 32 K-steps read h, rest read x
#define KTX_  (KT_ - KTH_)       // 8 K-steps from x
#define NBLK  64                 // persistent blocks; each owns NH hidden units
#define NH    16
#define WAVES 4
#define NTHR  (WAVES * 32)
#define LDS_HALVES (4 * KT_ * 32 * 16)   // 81920 bf16 = 160 KB weight slice
#define LDS_BYTES  (LDS_HALVES * 2)

typedef __attribute__((ext_vector_type(16))) __bf16        v16bf;
typedef __attribute__((ext_vector_type(8)))  float         v8f;
typedef __attribute__((ext_vector_type(4)))  unsigned int  v4u;
typedef int v4i __attribute__((vector_size(16)));          // matches async-builtin param

// explicit address spaces so A-loads / h-stores lower to global_*, not flat_*
typedef __attribute__((address_space(1))) unsigned short gu16;
typedef __attribute__((address_space(1))) float          gf32;

// -------- helpers --------
__device__ __forceinline__ unsigned short f2bf(float f) {
  unsigned int u = __float_as_uint(f);
  u += 0x7fffu + ((u >> 16) & 1u);       // round-to-nearest-even
  return (unsigned short)(u >> 16);
}
// branch-free activations: v_exp_f32 + v_rcp_f32, no exec-mask slow paths
__device__ __forceinline__ float fast_sigm(float x) {
  return __builtin_amdgcn_rcpf(1.0f + __expf(-x));
}
__device__ __forceinline__ float fast_tanh(float x) {
  float e = __expf(-2.0f * __builtin_fabsf(x));
  float t = (1.0f - e) * __builtin_amdgcn_rcpf(1.0f + e);
  return __builtin_copysignf(t, x);
}

// A-fragment (16x32 bf16): lane holds row M=lane&15, K chunks {c..c+7, c+16..c+23},
// c=(lane>>4)*8  ->  two 16B global loads 32B apart in a row-major [row][K] buffer.
__device__ __forceinline__ v16bf load_a_g(const gu16* p0) {
  union { v16bf v; v4u q[2]; } f;
  f.q[0] = *(const __attribute__((address_space(1))) v4u*)p0;
  f.q[1] = *(const __attribute__((address_space(1))) v4u*)(p0 + 16);
  return f.v;
}

// -------- grid-wide barrier (persistent kernel, NBLK blocks) --------
__device__ __forceinline__ void gridbar(unsigned int* sync) {
  __threadfence();
  __syncthreads();
  if (threadIdx.x == 0) {
    unsigned int g = __hip_atomic_load(&sync[1], __ATOMIC_ACQUIRE, __HIP_MEMORY_SCOPE_AGENT);
    unsigned int a = __hip_atomic_fetch_add(&sync[0], 1u, __ATOMIC_ACQ_REL, __HIP_MEMORY_SCOPE_AGENT);
    if (a == (unsigned)(NBLK - 1)) {
      __hip_atomic_store(&sync[0], 0u, __ATOMIC_RELAXED, __HIP_MEMORY_SCOPE_AGENT);
      __hip_atomic_fetch_add(&sync[1], 1u, __ATOMIC_ACQ_REL, __HIP_MEMORY_SCOPE_AGENT);
    } else {
      while (__hip_atomic_load(&sync[1], __ATOMIC_ACQUIRE, __HIP_MEMORY_SCOPE_AGENT) == g)
        __builtin_amdgcn_s_sleep(1);
    }
  }
  __syncthreads();
  __threadfence();
}

// -------- prep: combined weights -> bf16, WMMA B-fragment order --------
// Wfrag[b][g][kt][lane][e]: k = kt*32 + (lane&16?8:0) + (e<8 ? e : e+8) ; n = g*H + b*NH + (lane&15)
__global__ void lstm_prep_w(const float* __restrict__ Wih, const float* __restrict__ Whh,
                            unsigned short* __restrict__ wfrag) {
  int t = blockIdx.x * blockDim.x + threadIdx.x;         // 64*4*40*32 = 327680 lane-rows
  if (t >= NBLK * 4 * KT_ * 32) return;
  int lane = t & 31;
  int kt   = (t >> 5) % KT_;
  int g    = ((t >> 5) / KT_) & 3;
  int b    = (t >> 5) / (KT_ * 4);
  int n    = g * H_ + b * NH + (lane & 15);
  int kb   = kt * 32 + ((lane & 16) ? 8 : 0);
  __attribute__((aligned(16))) unsigned short tmp[16];
#pragma unroll
  for (int e = 0; e < 16; ++e) {
    int k = kb + (e < 8 ? e : e + 8);
    float w = (k < H_) ? Whh[(size_t)n * H_ + k] : Wih[(size_t)n * D_ + (k - H_)];
    tmp[e] = f2bf(w);
  }
  v4u* dst = (v4u*)(wfrag + (size_t)t * 16);
  dst[0] = ((const v4u*)tmp)[0];
  dst[1] = ((const v4u*)tmp)[1];
}

// -------- prep: x [B,S,D] f32 -> xbf [S][B][D] bf16 --------
__global__ void lstm_prep_x(const float* __restrict__ x, unsigned short* __restrict__ xbf) {
  int t = blockIdx.x * blockDim.x + threadIdx.x;          // 512*64*32 = 1048576
  int d8 = t & 31;
  int m  = (t >> 5) & 63;
  int s  = t >> 11;
  if (s >= S_) return;
  const float* src = x + ((size_t)m * S_ + s) * D_ + d8 * 8;
  __attribute__((aligned(16))) unsigned short tmp[8];
#pragma unroll
  for (int e = 0; e < 8; ++e) tmp[e] = f2bf(src[e]);
  *(v4u*)(xbf + ((size_t)s * B_ + m) * D_ + d8 * 8) = *(const v4u*)tmp;
}

// -------- prep: bias sum, zero h0, zero sync --------
__global__ void lstm_prep_misc(const float* __restrict__ bih, const float* __restrict__ bhh,
                               float* __restrict__ bsum, unsigned short* __restrict__ hbf0,
                               unsigned int* __restrict__ sync) {
  int t = blockIdx.x * blockDim.x + threadIdx.x;          // 32768 threads
  if (t < (B_ * H_) / 2) ((unsigned int*)hbf0)[t] = 0u;   // zero 128 KB h0
  if (t < 4 * H_) bsum[t] = bih[t] + bhh[t];
  if (t < 2) sync[t] = 0u;
}

// -------- one LSTM timestep (fully inlined; ping-pong handled by caller) --------
__device__ __forceinline__ void lstm_step(
    const unsigned short* __restrict__ smem, int lane,
    const gu16* __restrict__ hprev, gu16* __restrict__ hnext,
    const gu16* __restrict__ xs, gf32* __restrict__ hlast, bool write_last,
    int arow, int csel, int m0, int ncol, const float* biasv, v8f& cst) {
  v8f acc[4];
#pragma unroll
  for (int g = 0; g < 4; ++g)
#pragma unroll
    for (int e = 0; e < 8; ++e) acc[g][e] = biasv[g];

  // K-steps over h (global_load_b128 x2 per A-frag, ds_load_b128 x2 per B-frag)
  const gu16* ap = hprev + (size_t)arow * H_ + csel;
  for (int kt = 0; kt < KTH_; ++kt) {
    v16bf a = load_a_g(ap + kt * 32);
#pragma unroll
    for (int g = 0; g < 4; ++g) {
      const v16bf b = *(const v16bf*)(smem + ((size_t)(g * KT_ + kt) * 32 + lane) * 16);
      acc[g] = __builtin_amdgcn_wmma_f32_16x16x32_bf16(
                   false, a, false, b, (short)0, acc[g], false, false);
    }
  }
  // K-steps over x_t
  const gu16* xp = xs + (size_t)arow * D_ + csel;
#pragma unroll
  for (int kt = 0; kt < KTX_; ++kt) {
    v16bf a = load_a_g(xp + kt * 32);
#pragma unroll
    for (int g = 0; g < 4; ++g) {
      const v16bf b =
          *(const v16bf*)(smem + ((size_t)(g * KT_ + KTH_ + kt) * 32 + lane) * 16);
      acc[g] = __builtin_amdgcn_wmma_f32_16x16x32_bf16(
                   false, a, false, b, (short)0, acc[g], false, false);
    }
  }

  // i,f,g,o are lane-aligned across the four fragments -> pure per-lane update
#pragma unroll
  for (int e = 0; e < 8; ++e) {
    float iv = fast_sigm(acc[0][e]);
    float fv = fast_sigm(acc[1][e]);
    float gv = fast_tanh(acc[2][e]);
    float ov = fast_sigm(acc[3][e]);
    float cv = fv * cst[e] + iv * gv;
    cst[e] = cv;
    float hv = ov * fast_tanh(cv);
    hnext[(size_t)(m0 + e) * H_ + ncol] = f2bf(hv);
    if (write_last) hlast[(size_t)(m0 + e) * H_ + ncol] = hv;
  }
}

// -------- persistent recurrent kernel --------
__global__ void __launch_bounds__(NTHR, 1)
lstm_recurrent(const unsigned short* __restrict__ wfrag,
               const unsigned short* __restrict__ xbf,
               const float* __restrict__ bsum,
               unsigned short* __restrict__ hbf0,
               unsigned short* __restrict__ hbf1,
               float* __restrict__ hlast,
               unsigned int* __restrict__ sync) {
  extern __shared__ unsigned short smem[];
  const int tid   = threadIdx.x;
  const int w     = tid >> 5;               // wave id == batch row-tile mt
  const int lane  = tid & 31;
  const int nb    = blockIdx.x;             // hidden chunk (16 units)
  const int nlo   = lane & 15;
  const int khalf = (lane & 16) ? 1 : 0;

  const gu16* wfragg = (const gu16*)wfrag;
  const gu16* xg     = (const gu16*)xbf;
  gu16*       h0g    = (gu16*)hbf0;
  gu16*       h1g    = (gu16*)hbf1;
  gf32*       hlg    = (gf32*)hlast;

  // ---- pin this block's bf16 weight slice in LDS (async-copy path if available) ----
#if __has_builtin(__builtin_amdgcn_global_load_async_to_lds_b128) && \
    __has_builtin(__builtin_amdgcn_s_wait_asynccnt)
  {
    const gu16* src = wfragg + (size_t)nb * LDS_HALVES;
    for (int i = tid; i < LDS_HALVES / 8; i += NTHR)
      __builtin_amdgcn_global_load_async_to_lds_b128(
          (__attribute__((address_space(1))) v4i*)(src + (size_t)i * 8),
          (__attribute__((address_space(3))) v4i*)(smem + (size_t)i * 8), 0, 0);
    __builtin_amdgcn_s_wait_asynccnt(0);
  }
#else
  {
    const __attribute__((address_space(1))) v4u* src =
        (const __attribute__((address_space(1))) v4u*)(wfragg + (size_t)nb * LDS_HALVES);
    v4u* dst = (v4u*)smem;
    for (int i = tid; i < LDS_HALVES / 8; i += NTHR) dst[i] = src[i];
  }
#endif
  __syncthreads();

  float biasv[4];
#pragma unroll
  for (int g = 0; g < 4; ++g) biasv[g] = bsum[g * H_ + nb * NH + nlo];

  const int arow = w * 16 + nlo;            // A row this lane reads
  const int csel = khalf * 8;               // A k-chunk select within 32
  const int m0   = w * 16 + khalf * 8;      // first batch row of D-frag elements
  const int ncol = nb * NH + nlo;           // hidden column this lane owns

  v8f cst = {0.f, 0.f, 0.f, 0.f, 0.f, 0.f, 0.f, 0.f};   // cell state, register-resident

  for (int s = 0; s < S_; s += 2) {         // explicit ping-pong: no pointer selects
    lstm_step(smem, lane, h0g, h1g, xg + (size_t)s * (B_ * D_), hlg, false,
              arow, csel, m0, ncol, biasv, cst);
    __builtin_prefetch(xbf + (size_t)(s + 1) * (B_ * D_) + tid * ((B_ * D_) / NTHR), 0, 0);
    gridbar(sync);
    lstm_step(smem, lane, h1g, h0g, xg + (size_t)(s + 1) * (B_ * D_), hlg, (s + 1 == S_ - 1),
              arow, csel, m0, ncol, biasv, cst);
    if (s + 2 < S_)
      __builtin_prefetch(xbf + (size_t)(s + 2) * (B_ * D_) + tid * ((B_ * D_) / NTHR), 0, 0);
    gridbar(sync);
  }
}

// -------- final FC: out[64][128] = hlast @ Wfc^T + bfc (tiny) --------
__global__ void lstm_fc(const float* __restrict__ hlast, const float* __restrict__ Wfc,
                        const float* __restrict__ bfc, float* __restrict__ out) {
  int t = blockIdx.x * blockDim.x + threadIdx.x;
  if (t >= B_ * O_) return;
  int o = t & (O_ - 1);
  int m = t >> 7;
  const gf32* hp = (const gf32*)(hlast + (size_t)m * H_);
  const gf32* wp = (const gf32*)(Wfc + (size_t)o * H_);
  float acc = bfc[o];
#pragma unroll 4
  for (int k = 0; k < H_; ++k) acc = fmaf(hp[k], wp[k], acc);
  ((gf32*)out)[(size_t)m * O_ + o] = acc;
}

// -------- workspace layout (bytes, 256-aligned) : total ~26.6 MB --------
#define OFF_WFRAG 0u
#define SZ_WFRAG  ((size_t)NBLK * LDS_BYTES)                 // 10,485,760
#define OFF_XBF   (OFF_WFRAG + SZ_WFRAG)
#define SZ_XBF    ((size_t)S_ * B_ * D_ * 2)                 // 16,777,216
#define OFF_HBF0  (OFF_XBF + SZ_XBF)
#define SZ_HBF    ((size_t)B_ * H_ * 2)                      // 131,072
#define OFF_HBF1  (OFF_HBF0 + SZ_HBF)
#define OFF_HLAST (OFF_HBF1 + SZ_HBF)
#define SZ_HLAST  ((size_t)B_ * H_ * 4)                      // 262,144
#define OFF_BSUM  (OFF_HLAST + SZ_HLAST)
#define SZ_BSUM   ((size_t)4 * H_ * 4)                       // 16,384
#define OFF_SYNC  (OFF_BSUM + SZ_BSUM)

extern "C" void kernel_launch(void* const* d_in, const int* in_sizes, int n_in,
                              void* d_out, int out_size, void* d_ws, size_t ws_size,
                              hipStream_t stream) {
  (void)in_sizes; (void)n_in; (void)out_size; (void)ws_size;
  const float* x   = (const float*)d_in[0];
  const float* Wih = (const float*)d_in[1];
  const float* Whh = (const float*)d_in[2];
  const float* bih = (const float*)d_in[3];
  const float* bhh = (const float*)d_in[4];
  const float* Wfc = (const float*)d_in[5];
  const float* bfc = (const float*)d_in[6];

  char* ws = (char*)d_ws;
  unsigned short* wfrag = (unsigned short*)(ws + OFF_WFRAG);
  unsigned short* xbf   = (unsigned short*)(ws + OFF_XBF);
  unsigned short* hbf0  = (unsigned short*)(ws + OFF_HBF0);
  unsigned short* hbf1  = (unsigned short*)(ws + OFF_HBF1);
  float*          hlast = (float*)(ws + OFF_HLAST);
  float*          bsum  = (float*)(ws + OFF_BSUM);
  unsigned int*   sync  = (unsigned int*)(ws + OFF_SYNC);

  (void)hipFuncSetAttribute((const void*)lstm_recurrent,
                            hipFuncAttributeMaxDynamicSharedMemorySize, LDS_BYTES);

  lstm_prep_w   <<<(NBLK * 4 * KT_ * 32 + 255) / 256, 256, 0, stream>>>(Wih, Whh, wfrag);
  lstm_prep_x   <<<(S_ * B_ * (D_ / 8) + 255) / 256, 256, 0, stream>>>(x, xbf);
  lstm_prep_misc<<<128, 256, 0, stream>>>(bih, bhh, bsum, hbf0, sync);
  lstm_recurrent<<<NBLK, NTHR, LDS_BYTES, stream>>>(wfrag, xbf, bsum, hbf0, hbf1, hlast, sync);
  lstm_fc       <<<(B_ * O_ + 255) / 256, 256, 0, stream>>>(hlast, Wfc, bfc, (float*)d_out);
}
